// SPDNormalization_79697413144765
// MI455X (gfx1250) — matrix operand order repaired and checked by reading, not compiled, not run on hardware
//
#include <hip/hip_runtime.h>
#include <hip/hip_bf16.h>

// ============================================================================
// SPDNormalization forward for MI455X (gfx1250, wave32, WMMA + TDM).
//
// Matmul-only matrix functions (all FLOPs land on v_wmma_f32_16x16x32_bf16):
//   * A^{1/2}, A^{-1/2}: trace-scale (eig in (0,1]) + coupled Newton-Schulz
//   * logm: inverse scaling-and-squaring (4 NS roots) + 4-term series
//   * expm: scaling-and-squaring (2^-6) + 8-term Horner Taylor
//   * out_i = W x_i W^T,  W = sqrtm(G) @ inv_sqrtm(center)
//
// Data movement:
//   * 64x64 f32 tiles pulled global->LDS by the Tensor Data Mover
//     (tensor_load_to_lds, D# pad fields create the pitch-68 layout in HW),
//     waited with s_wait_tensorcnt. Fallback: float4 copies.
//   * Per-GEMM cooperative bf16 fragment staging: operands pre-swizzled once
//     into WMMA lane layout; each lane then loads its fragment as 32
//     contiguous bytes (2x ds_load_b128) instead of 8 scattered gathers.
//   * f32 master matrices padded to pitch 68 -> conflict-free staging reads.
// ============================================================================

typedef __attribute__((ext_vector_type(16))) __bf16 v16bf;
typedef __attribute__((ext_vector_type(8)))  float  v8f;
typedef unsigned int u32x4 __attribute__((ext_vector_type(4)));
typedef int          i32x8 __attribute__((ext_vector_type(8)));
typedef int          i32x4 __attribute__((ext_vector_type(4)));

#define MATN   64
#define MATSZ  4096            // logical 64*64
#define PITCH  68              // padded row pitch (floats): bank-conflict free
#define LSLOT  (MATN * PITCH)  // 4352 floats per padded LDS matrix slot
#define NTHR   256             // 8 waves of 32
#define SAMP   8               // samples per persistent block

#if __has_builtin(__builtin_amdgcn_tensor_load_to_lds) && \
    __has_builtin(__builtin_amdgcn_s_wait_tensorcnt)
#define USE_TDM 1
#else
#define USE_TDM 0
#endif

__device__ __forceinline__ int pidx(int i) { return (i >> 6) * PITCH + (i & 63); }

// ---------------------------------------------------------------------------
// Workgroup context: padded f32 slots + shared bf16 fragment staging buffers.
// ---------------------------------------------------------------------------
struct WgCtx {
  float*  lds;   // base of padded f32 slots
  __bf16* fa;    // 4096 bf16: 8 A fragments (4 tile-rows x 2 k-blocks)
  __bf16* fb;    // 4096 bf16: 8 B fragments (4 tile-cols x 2 k-blocks)
  float*  red;   // 64-float reduction scratch
};

__device__ __forceinline__ WgCtx make_ctx(float* lds, int nslots) {
  WgCtx c;
  c.lds = lds;
  c.fa  = (__bf16*)(lds + (size_t)nslots * LSLOT);  // 32B-aligned by layout
  c.fb  = c.fa + MATSZ;
  c.red = (float*)(c.fa + 2 * MATSZ);
  return c;
}

#define LB(i) (ctx.lds + (size_t)(i) * LSLOT)

// ---------------------------------------------------------------------------
// Global -> padded LDS 64x64 f32 tile load via Tensor Data Mover.
// D#: 2D tensor, data_size=4B, tile 64x64, stride 64, LDS padding of
// 4 dwords every 64 dwords -> pitch-68 layout written by hardware.
// ---------------------------------------------------------------------------
__device__ __forceinline__ void wg_load_tile(const WgCtx& ctx, float* slot,
                                             const float* __restrict__ src) {
#if USE_TDM
  if (threadIdx.x < 32) {  // TDM is a per-wave DMA: issue from wave 0 only
    const unsigned long long ga = (unsigned long long)(const void*)src;
    // dynamic LDS starts at offset 0 (no static __shared__ in this TU)
    const unsigned ldsAddr = (unsigned)((const char*)slot - (const char*)ctx.lds);
    u32x4 g0;
    g0[0] = 1u;                                   // count=1, user descriptor
    g0[1] = ldsAddr;                              // lds_addr (bytes)
    g0[2] = (unsigned)(ga & 0xffffffffu);         // global_addr[31:0]
    g0[3] = (unsigned)((ga >> 32) & 0x01ffffffu)  // global_addr[56:32]
            | (2u << 30);                         // type = 2 ("image")
    i32x8 g1;
    g1[0] = (2 << 16)      // data_size = 4 bytes
          | (1 << 20)      // pad_enable
          | (5 << 22)      // pad_interval: 2^(5+1) = 64 dwords
          | (3 << 25);     // pad_amount:   3+1    = 4 dwords
    g1[1] = (MATN << 16);  // tensor_dim0 = 64 (low 16 @ bits 63:48)
    g1[2] = (MATN << 16);  // tensor_dim1 = 64 (low 16 @ bits 95:80)
    g1[3] = (MATN << 16);  // tile_dim0   = 64 (bits 127:112)
    g1[4] = MATN;          // tile_dim1   = 64 (bits 143:128)
    g1[5] = MATN;          // tensor_dim0_stride = 64 (bits 207:160)
    g1[6] = 0;
    g1[7] = 0;
    const i32x4 z4 = {0, 0, 0, 0};                // 2D: groups 2/3 unused
    const i32x8 z8 = {0, 0, 0, 0, 0, 0, 0, 0};    // unused extra group
    __builtin_amdgcn_tensor_load_to_lds(g0, g1, z4, z4, z8, 0);
    __builtin_amdgcn_s_wait_tensorcnt(0);
  }
  __syncthreads();
#else
  const float4* s4 = (const float4*)src;
  float4*       d4 = (float4*)slot;
  for (int i = threadIdx.x; i < MATSZ / 4; i += NTHR)
    d4[(i >> 4) * (PITCH / 4) + (i & 15)] = s4[i];
  __syncthreads();
#endif
}

// padded LDS slot -> dense global
__device__ __forceinline__ void wg_s2g(float* __restrict__ dst, const float* slot) {
  float4*       d4 = (float4*)dst;
  const float4* s4 = (const float4*)slot;
  for (int i = threadIdx.x; i < MATSZ / 4; i += NTHR)
    d4[i] = s4[(i >> 4) * (PITCH / 4) + (i & 15)];
}

// ---------------------------------------------------------------------------
// Workgroup 64x64x64 GEMM: C = A * B (or A * B^T).  Stage 1: cooperative
// f32->bf16 conversion into WMMA fragment layout (each operand converted
// exactly once).  Stage 2: 16 C-tiles, 2 per wave, K=64 as 2 WMMA steps;
// fragments load as contiguous 32B per lane.
// ---------------------------------------------------------------------------
__device__ __forceinline__ void wg_mm64(const WgCtx& ctx,
                                        const float* __restrict__ A,
                                        const float* __restrict__ B,
                                        float* __restrict__ C,
                                        int transB) {
  __syncthreads();                        // inputs fully produced
  const int tid  = threadIdx.x;
  const int lane = tid & 31;
  const int half = lane >> 4;             // 0/1
  const int l16  = lane & 15;
  {
    // ---- stage fragments: thread (f = tid>>5, lane) owns one 16-elem slot
    const int f  = tid >> 5;              // fragment id 0..7
    const int kb = (f & 1) * 32;          // k-block base
    // A fragment (16x32) for tile-row f>>1:
    //   e<8 -> K = kb+base0+e ; e>=8 -> K = kb+base0+16+(e-8)
    {
      const int    row   = (f >> 1) * 16 + l16;
      const int    base0 = half * 8;
      const float* Ar    = A + row * PITCH + kb + base0;
      __bf16*      dst   = ctx.fa + ((f * 32 + lane) * 16);
      #pragma unroll
      for (int e = 0; e < 8; ++e)  dst[e] = (__bf16)Ar[e];
      #pragma unroll
      for (int e = 8; e < 16; ++e) dst[e] = (__bf16)Ar[e + 8];
    }
    // B fragment (32x16) for tile-col f>>1:  K = kb + half*8 + e (+8 if e>=8)
    {
      const int col   = (f >> 1) * 16 + l16;
      const int kbase = kb + half * 8;
      __bf16*   dst   = ctx.fb + ((f * 32 + lane) * 16);
      #pragma unroll
      for (int e = 0; e < 16; ++e) {
        const int kk = kbase + (e < 8 ? e : e + 8);
        const float v = transB ? B[col * PITCH + kk] : B[kk * PITCH + col];
        dst[e] = (__bf16)v;
      }
    }
  }
  __syncthreads();                        // fragments staged
  {
    const int wv = tid >> 5;
    #pragma unroll
    for (int tt = 0; tt < 2; ++tt) {
      const int t  = wv * 2 + tt;         // 0..15
      const int ti = t >> 2;              // tile row
      const int tj = t & 3;               // tile col
      v8f acc = {0.f, 0.f, 0.f, 0.f, 0.f, 0.f, 0.f, 0.f};
      #pragma unroll
      for (int kbi = 0; kbi < 2; ++kbi) {
        const v16bf a = *(const v16bf*)(ctx.fa + (((ti * 2 + kbi) * 32 + lane) * 16));
        const v16bf b = *(const v16bf*)(ctx.fb + (((tj * 2 + kbi) * 32 + lane) * 16));
        acc = __builtin_amdgcn_wmma_f32_16x16x32_bf16(
                  false, a, false, b, (short)0, acc, false, false);
      }
      #pragma unroll
      for (int r = 0; r < 8; ++r)
        C[(ti * 16 + half * 8 + r) * PITCH + tj * 16 + l16] = acc[r];
    }
  }
  __syncthreads();                        // C fully produced
}

// ------------------------- elementwise helpers -----------------------------
__device__ __forceinline__ void wg_scale_diag(float* dst, const float* src,
                                              float alpha, float beta) {
  for (int i = threadIdx.x; i < MATSZ; i += NTHR) {
    float v = alpha * src[pidx(i)];
    if ((i >> 6) == (i & 63)) v += beta;
    dst[pidx(i)] = v;
  }
  __syncthreads();
}

__device__ __forceinline__ void wg_set_identity(float* dst) {
  for (int i = threadIdx.x; i < MATSZ; i += NTHR)
    dst[pidx(i)] = ((i >> 6) == (i & 63)) ? 1.f : 0.f;
  __syncthreads();
}

__device__ __forceinline__ void wg_copy(float* dst, const float* src) {
  for (int i = threadIdx.x; i < MATSZ; i += NTHR) dst[pidx(i)] = src[pidx(i)];
  __syncthreads();
}

__device__ __forceinline__ float wg_trace(const WgCtx& ctx, const float* A) {
  if (threadIdx.x < MATN) ctx.red[threadIdx.x] = A[threadIdx.x * (PITCH + 1)];
  __syncthreads();
  if (threadIdx.x == 0) {
    float s = 0.f;
    for (int i = 0; i < MATN; ++i) s += ctx.red[i];
    ctx.red[0] = s;
  }
  __syncthreads();
  const float t = ctx.red[0];
  __syncthreads();
  return t;
}

__device__ __forceinline__ void free3(int used, int& a, int& b, int& c) {
  int s[3], n = 0;
  for (int i = 1; i <= 4; ++i) if (i != used) s[n++] = i;
  a = s[0]; b = s[1]; c = s[2];
}

// Coupled Newton-Schulz on slot iA (eig in (0,1]); Y ~ A^{1/2} -> *oy,
// Z ~ A^{-1/2} -> *oz. Uses slots {iA,f1,f2,f3}.
__device__ __forceinline__ void ns_pair(const WgCtx& ctx, int iA, int f1,
                                        int f2, int f3, int iters,
                                        int& oy, int& oz) {
  int iY = iA, iZ = f1, t0 = f2, sp = f3;
  wg_set_identity(LB(iZ));
  for (int it = 0; it < iters; ++it) {
    wg_mm64(ctx, LB(iZ), LB(iY), LB(t0), 0);        // T = Z*Y
    wg_scale_diag(LB(t0), LB(t0), -0.5f, 1.5f);     // T = 1.5I - 0.5T
    wg_mm64(ctx, LB(iY), LB(t0), LB(sp), 0);        // Ynew = Y*T
    wg_mm64(ctx, LB(t0), LB(iZ), LB(iY), 0);        // Znew = T*Z (old Y slot)
    const int oldY = iY, oldZ = iZ;
    iY = sp; iZ = oldY; sp = oldZ;                  // t0 stays the T slot
  }
  oy = iY; oz = iZ;
}

// ---------------------------------------------------------------------------
// Kernel 1: zero scratch (mean + tangent partials).
// ---------------------------------------------------------------------------
__global__ void __launch_bounds__(NTHR)
k_spdn_zero(float* __restrict__ p, int n) {
  const int i = blockIdx.x * NTHR + threadIdx.x;
  if (i < n) p[i] = 0.f;
}

// ---------------------------------------------------------------------------
// Kernel 2: arithmetic batch mean (128 MB read @ 23.3 TB/s ~ 5.5 us).
// ---------------------------------------------------------------------------
__global__ void __launch_bounds__(NTHR)
k_spdn_mean(const float* __restrict__ x, float* __restrict__ mean, int Bn) {
  const int    col  = blockIdx.x * NTHR + threadIdx.x;     // 0..4095
  const size_t base = (size_t)blockIdx.y * 128 * MATSZ;
  float acc = 0.f;
  for (int b = 0; b < 128; ++b)
    acc += x[base + (size_t)b * MATSZ + col];
  atomicAdd(&mean[col], acc * (1.0f / (float)Bn));
}

// ---------------------------------------------------------------------------
// Kernel 3: s = sqrtm(mean), si = inv_sqrtm(mean). Single workgroup.
// ---------------------------------------------------------------------------
__global__ void __launch_bounds__(NTHR)
k_spdn_mean_sqrt(const float* __restrict__ mean, float* __restrict__ sOut,
                 float* __restrict__ siOut) {
  extern __shared__ float ldsraw[];
  WgCtx ctx = make_ctx(ldsraw, 5);
  wg_load_tile(ctx, LB(1), mean);
  float c = fmaxf(wg_trace(ctx, LB(1)), 1e-8f);
  wg_scale_diag(LB(1), LB(1), 1.f / c, 0.f);
  int oy, oz;
  ns_pair(ctx, 1, 2, 3, 4, 12, oy, oz);
  const float sc = sqrtf(c), isc = rsqrtf(c);
  for (int i = threadIdx.x; i < MATSZ; i += NTHR) {
    sOut[i]  = LB(oy)[pidx(i)] * sc;
    siOut[i] = LB(oz)[pidx(i)] * isc;
  }
}

// ---------------------------------------------------------------------------
// Kernel 4: Karcher tangent step (FLOP-dominant). Persistent: SAMP samples
// per block, si loaded once. Per sample: M = si x si; Tpart += logm(M).
// ---------------------------------------------------------------------------
__global__ void __launch_bounds__(NTHR)
k_spdn_karcher(const float* __restrict__ x, const float* __restrict__ si,
               float* __restrict__ Tpart, int kslots) {
  extern __shared__ float ldsraw[];
  WgCtx ctx = make_ctx(ldsraw, 5);
  wg_load_tile(ctx, LB(0), si);                     // si persists in slot 0

  for (int s = 0; s < SAMP; ++s) {
    const int idx = blockIdx.x * SAMP + s;
    wg_load_tile(ctx, LB(1), x + (size_t)idx * MATSZ);
    wg_mm64(ctx, LB(0), LB(1), LB(2), 0);           // t1 = si * x
    wg_mm64(ctx, LB(2), LB(0), LB(1), 0);           // M  = t1 * si

    float c = fmaxf(wg_trace(ctx, LB(1)), 1e-8f);
    wg_scale_diag(LB(1), LB(1), 1.f / c, 0.f);      // A0 = M/c, eig in (0,1]

    int iA = 1;
    for (int r = 0; r < 4; ++r) {                   // A <- A^{1/2}, 4 times
      int f1, f2, f3, oy, oz;
      free3(iA, f1, f2, f3);
      ns_pair(ctx, iA, f1, f2, f3, 7, oy, oz);
      iA = oy;
    }
    wg_scale_diag(LB(iA), LB(iA), 1.f, -1.f);       // E = A - I
    int g1, g2, g3;
    free3(iA, g1, g2, g3);
    wg_mm64(ctx, LB(iA), LB(iA), LB(g1), 0);        // E^2
    wg_mm64(ctx, LB(g1), LB(iA), LB(g2), 0);        // E^3
    wg_mm64(ctx, LB(g1), LB(g1), LB(g3), 0);        // E^4

    const float  lc = logf(c);
    float*       Tp = Tpart + (size_t)(idx % kslots) * MATSZ;
    const float *E = LB(iA), *M2 = LB(g1), *M3 = LB(g2), *M4 = LB(g3);
    for (int i = threadIdx.x; i < MATSZ; i += NTHR) {
      const int a = pidx(i);
      float v = 16.f * (E[a] - 0.5f * M2[a] + (1.f / 3.f) * M3[a] - 0.25f * M4[a]);
      if ((i >> 6) == (i & 63)) v += lc;
      atomicAdd(&Tp[i], v);
    }
    __syncthreads();
  }
}

// ---------------------------------------------------------------------------
// Kernel 5: finalize: expm(Tbar), center = s expT s, csi, g_sqrt, W.
// ---------------------------------------------------------------------------
__global__ void __launch_bounds__(NTHR)
k_spdn_finalize(const float* __restrict__ Tpart, const float* __restrict__ sM,
                const float* __restrict__ G, float* __restrict__ wsW,
                float* __restrict__ wsCsi, int Bn, int kslots) {
  extern __shared__ float ldsraw[];
  WgCtx ctx = make_ctx(ldsraw, 5);

  // Es = Tbar / 2^6 = (sum_k Tpart) / (Bn * 64)
  {
    const float sc = 1.f / ((float)Bn * 64.f);
    for (int i = threadIdx.x; i < MATSZ; i += NTHR) {
      float a = 0.f;
      for (int k = 0; k < kslots; ++k) a += Tpart[(size_t)k * MATSZ + i];
      LB(1)[pidx(i)] = a * sc;
    }
    __syncthreads();
  }
  // exp(Es): Horner Taylor, 8 terms (R in slot 2)
  wg_scale_diag(LB(2), LB(1), 1.f / 8.f, 1.f);      // R = I + Es/8
  for (int j = 7; j >= 1; --j) {
    wg_mm64(ctx, LB(1), LB(2), LB(3), 0);
    wg_scale_diag(LB(2), LB(3), 1.f / (float)j, 1.f);
  }
  for (int sq = 0; sq < 6; ++sq) {                  // undo 2^-6 scaling
    wg_mm64(ctx, LB(2), LB(2), LB(3), 0);
    wg_copy(LB(2), LB(3));
  }
  // center = s * expT * s
  wg_load_tile(ctx, LB(1), sM);
  wg_mm64(ctx, LB(1), LB(2), LB(3), 0);
  wg_mm64(ctx, LB(3), LB(1), LB(4), 0);             // center in slot 4
  // csi = inv_sqrtm(center)
  float c2 = fmaxf(wg_trace(ctx, LB(4)), 1e-8f);
  wg_scale_diag(LB(4), LB(4), 1.f / c2, 0.f);
  int oy, oz;
  ns_pair(ctx, 4, 1, 2, 3, 12, oy, oz);
  {
    const float isc = rsqrtf(c2);
    for (int i = threadIdx.x; i < MATSZ; i += NTHR)
      wsCsi[i] = LB(oz)[pidx(i)] * isc;
    __syncthreads();
  }
  // g_sqrt = sqrtm(G)
  wg_load_tile(ctx, LB(1), G);
  float c3 = fmaxf(wg_trace(ctx, LB(1)), 1e-8f);
  wg_scale_diag(LB(1), LB(1), 1.f / c3, 0.f);
  ns_pair(ctx, 1, 2, 3, 4, 12, oy, oz);
  wg_scale_diag(LB(oy), LB(oy), sqrtf(c3), 0.f);    // g_sqrt in slot oy
  // W = g_sqrt * csi
  int f1, f2, f3;
  free3(oy, f1, f2, f3);
  wg_load_tile(ctx, LB(f1), wsCsi);
  wg_mm64(ctx, LB(oy), LB(f1), LB(f2), 0);
  wg_s2g(wsW, LB(f2));
}

// ---------------------------------------------------------------------------
// Kernel 6: re-color: out_i = W * x_i * W^T. Persistent: SAMP samples per
// block, W loaded once (removes 128 MB of redundant W re-reads).
// ---------------------------------------------------------------------------
__global__ void __launch_bounds__(NTHR)
k_spdn_recolor(const float* __restrict__ x, const float* __restrict__ W,
               float* __restrict__ out) {
  extern __shared__ float ldsraw[];
  WgCtx ctx = make_ctx(ldsraw, 3);
  float* bW = LB(0);
  float* bX = LB(1);
  float* bT = LB(2);
  wg_load_tile(ctx, bW, W);
  for (int s = 0; s < SAMP; ++s) {
    const size_t idx = (size_t)blockIdx.x * SAMP + s;
    wg_load_tile(ctx, bX, x + idx * MATSZ);
    wg_mm64(ctx, bW, bX, bT, 0);                    // T = W * x
    wg_mm64(ctx, bT, bW, bX, 1);                    // out = T * W^T
    wg_s2g(out + idx * MATSZ, bX);
    __syncthreads();                                // bX reload hazard
  }
}

// ---------------------------------------------------------------------------
// Host launcher
// ---------------------------------------------------------------------------
extern "C" void kernel_launch(void* const* d_in, const int* in_sizes, int n_in,
                              void* d_out, int out_size, void* d_ws, size_t ws_size,
                              hipStream_t stream) {
  (void)n_in; (void)out_size; (void)ws_size;
  const float* x = (const float*)d_in[0];            // [B,64,64] f32
  const float* G = (const float*)d_in[1];            // [64,64]   f32
  float* out = (float*)d_out;                        // [B,64,64] f32
  float* ws  = (float*)d_ws;

  const int Bn     = in_sizes[0] / MATSZ;            // 8192
  const int KSLOTS = 64;

  // ws layout (floats): [mean 4096][Tpart 64*4096][s][si][csi][W]
  float* mean  = ws;
  float* Tpart = ws + MATSZ;
  float* sM    = ws + MATSZ + KSLOTS * MATSZ;
  float* siM   = sM  + MATSZ;
  float* csi   = siM + MATSZ;
  float* W     = csi + MATSZ;

  // shared: padded f32 slots + bf16 frag buffers (16 KB) + reduce scratch
  const size_t shPool = (size_t)(5 * LSLOT) * sizeof(float) + 2 * MATSZ * 2 + 256;
  const size_t shRec  = (size_t)(3 * LSLOT) * sizeof(float) + 2 * MATSZ * 2 + 256;

  const int nz = MATSZ + KSLOTS * MATSZ;             // mean + Tpart
  k_spdn_zero<<<(nz + NTHR - 1) / NTHR, NTHR, 0, stream>>>(ws, nz);

  dim3 gmean(16, Bn / 128);
  k_spdn_mean<<<gmean, NTHR, 0, stream>>>(x, mean, Bn);

  k_spdn_mean_sqrt<<<1, NTHR, shPool, stream>>>(mean, sM, siM);

  k_spdn_karcher<<<Bn / SAMP, NTHR, shPool, stream>>>(x, siM, Tpart, KSLOTS);

  k_spdn_finalize<<<1, NTHR, shPool, stream>>>(Tpart, sM, G, W, csi, Bn, KSLOTS);

  k_spdn_recolor<<<Bn / SAMP, NTHR, shRec, stream>>>(x, W, out);
}
